// STCN_29875792511367
// MI455X (gfx1250) — matrix operand order repaired and checked by reading, not compiled, not run on hardware
//
#include <hip/hip_runtime.h>
#include <math.h>
#include <stdint.h>

typedef __attribute__((ext_vector_type(16))) _Float16 v16h;
typedef __attribute__((ext_vector_type(8)))  _Float16 h8;
typedef __attribute__((ext_vector_type(8)))  float    v8f;

// Problem constants
constexpr int Bn  = 4;
constexpr int CK  = 512;
constexpr int CV  = 512;
constexpr int Tt  = 3;
constexpr int Hh  = 24;
constexpr int Ww  = 24;
constexpr int THW = Tt * Hh * Ww;   // 1728
constexpr int HW  = Hh * Ww;        // 576

// ---------------------------------------------------------------------------
// Workspace layout
// ---------------------------------------------------------------------------
// f32 region (float elements)
constexpr size_t SZ_AFF    = (size_t)Bn * THW * HW;
constexpr size_t SZ_ROWSQ  = (size_t)Bn * THW;
constexpr size_t SZ_FUSION = (size_t)Bn * 1536 * HW;
constexpr size_t SZ_POOL   = (size_t)Bn * 1536;
constexpr size_t SZ_MBUF   = (size_t)Bn * 512 * HW;
constexpr size_t SZ_DEC    = (size_t)Bn * 1024 * HW;
constexpr size_t SZ_T1     = (size_t)Bn * 256 * 9216;
constexpr size_t SZ_T2     = (size_t)Bn * 256 * 9216;
constexpr size_t SZ_X16    = (size_t)Bn * 512 * HW;
constexpr size_t SZ_S8     = (size_t)Bn * 512 * 2304;
constexpr size_t SZ_X8     = (size_t)Bn * 256 * 2304;
constexpr size_t SZ_S4     = (size_t)Bn * 256 * 9216;
constexpr size_t SZ_PRED   = (size_t)Bn * 9216;

constexpr size_t OFF_AFF    = 0;
constexpr size_t OFF_ROWSQ  = OFF_AFF    + SZ_AFF;
constexpr size_t OFF_FUSION = OFF_ROWSQ  + SZ_ROWSQ;
constexpr size_t OFF_POOL   = OFF_FUSION + SZ_FUSION;
constexpr size_t OFF_MBUF   = OFF_POOL   + SZ_POOL;
constexpr size_t OFF_DEC    = OFF_MBUF   + SZ_MBUF;
constexpr size_t OFF_T1     = OFF_DEC    + SZ_DEC;
constexpr size_t OFF_T2     = OFF_T1     + SZ_T1;
constexpr size_t OFF_X16    = OFF_T2     + SZ_T2;
constexpr size_t OFF_S8     = OFF_X16    + SZ_X16;
constexpr size_t OFF_X8     = OFF_S8     + SZ_S8;
constexpr size_t OFF_S4     = OFF_X8     + SZ_X8;
constexpr size_t OFF_PRED   = OFF_S4     + SZ_S4;
constexpr size_t F32_TOTAL  = OFF_PRED   + SZ_PRED;   // multiple of 4 => 16B aligned

// f16 region (half elements), appended after the f32 region
constexpr size_t HZ_KT   = (size_t)Bn * THW * CK;   // mk^T  [B,THW,CK]
constexpr size_t HZ_QT   = (size_t)Bn * HW * CK;    // qk^T  [B,HW,CK]
constexpr size_t HZ_MVH  = (size_t)Bn * CV * THW;   // mv    [B,CV,THW]
constexpr size_t HZ_AFFT = (size_t)Bn * HW * THW;   // aff^T [B,HW,THW]
constexpr size_t HZ_WP   = (size_t)9 * 512 * 1536;  // packed weights (max)
constexpr size_t HZ_XT   = (size_t)Bn * 9216 * 256; // staged activations (max)

constexpr size_t HOFF_KT   = 0;
constexpr size_t HOFF_QT   = HOFF_KT   + HZ_KT;
constexpr size_t HOFF_MVH  = HOFF_QT   + HZ_QT;
constexpr size_t HOFF_AFFT = HOFF_MVH  + HZ_MVH;
constexpr size_t HOFF_WP   = HOFF_AFFT + HZ_AFFT;
constexpr size_t HOFF_XT1  = HOFF_WP   + HZ_WP;
constexpr size_t HOFF_XT2  = HOFF_XT1  + HZ_XT;

// ---------------------------------------------------------------------------
// CDNA5 helpers
// ---------------------------------------------------------------------------
__device__ __forceinline__ v16h cat16(h8 a, h8 b) {
    return __builtin_shufflevector(a, b, 0, 1, 2, 3, 4, 5, 6, 7,
                                   8, 9, 10, 11, 12, 13, 14, 15);
}

// Async copy 16B global -> LDS (GLOBAL_LOAD_ASYNC_TO_LDS_B128, tracked by ASYNCcnt)
__device__ __forceinline__ void async_ld_b128(uint32_t lds_off, const void* gaddr) {
    asm volatile("global_load_async_to_lds_b128 %0, %1, off"
                 :: "v"(lds_off), "v"(gaddr) : "memory");
}
__device__ __forceinline__ void wait_async0() {
    asm volatile("s_wait_asynccnt 0x0" ::: "memory");
}
// Each thread keeps exactly 4 chunks in flight per tile; async loads complete
// in order, so ASYNCcnt <= 4 retires precisely the previous tile.
__device__ __forceinline__ void wait_async4() {
    asm volatile("s_wait_asynccnt 0x4" ::: "memory");
}

// ---------------------------------------------------------------------------
// Tiled f16 GEMM: C[b] = A[b] (MxK, row-major) * BT[b]^T (BT is NxK row-major)
// 4 waves/block; 64x64 macro-tile; double-buffered LDS filled with
// global_load_async_to_lds_b128; 4 WMMA per wave per K-step.
// mode 0: affinity epilogue  out = (2*acc - rs[m]) / sqrt(CK)
// mode 1: plain epilogue     out[(b*Ctot + m)*N + n] = acc
// ---------------------------------------------------------------------------
__global__ void k_gemm64(const _Float16* __restrict__ A, const _Float16* __restrict__ BT,
                         const float* __restrict__ rs, float* __restrict__ out,
                         int M, int N, int K, int mode, int Ctot) {
    // buffer: [2] x (A tile 64x32 halves | B tile 64x32 halves) = 2 x 8KB
    __shared__ __align__(16) _Float16 lds[2][4096];
    int tid = threadIdx.x;               // 0..127
    int lane = tid & 31, w = tid >> 5;
    int hi = lane >> 4, lo = lane & 15;
    int n0 = blockIdx.x * 64, m0 = blockIdx.y * 64, b = blockIdx.z;
    const _Float16* Ab = A  + (size_t)b * M * K;
    const _Float16* Bb = BT + (size_t)b * N * K;
    uint32_t base0 = (uint32_t)(uintptr_t)&lds[0][0];
    uint32_t base1 = (uint32_t)(uintptr_t)&lds[1][0];
    v8f acc[4] = {{}, {}, {}, {}};

    // Uniform (branch-free) cooperative tile issue: 2 A-chunks + 2 B-chunks per thread.
    auto issue_tile = [&](int kk, uint32_t bufbase) {
#pragma unroll
        for (int j = 0; j < 2; j++) {
            int q = tid + j * 128;                 // A chunk id, 0..255
            int r = q >> 2, c = q & 3;
            async_ld_b128(bufbase + (uint32_t)(r * 64 + c * 16),
                          Ab + (size_t)(m0 + r) * K + kk + c * 8);
        }
#pragma unroll
        for (int j = 0; j < 2; j++) {
            int q = tid + j * 128;                 // B chunk id, 0..255
            int r = q >> 2, c = q & 3;
            async_ld_b128(bufbase + 4096u + (uint32_t)(r * 64 + c * 16),
                          Bb + (size_t)(n0 + r) * K + kk + c * 8);
        }
    };

    issue_tile(0, base0);
    int cur = 0;
    for (int k0 = 0; k0 < K; k0 += 32) {
        // Prefetch next tile into the other buffer, then wait only for current tile.
        if (k0 + 32 < K) {
            issue_tile(k0 + 32, cur ? base0 : base1);
            wait_async4();
        } else {
            wait_async0();
        }
        __syncthreads();   // all waves' chunks of current tile are in LDS

        const _Float16* lA = &lds[cur][0];
        const _Float16* lB = &lds[cur][2048];
        int ml = w * 16 + lo;
        h8 a0 = *(const h8*)&lA[ml * 32 + hi * 8];
        h8 a1 = *(const h8*)&lA[ml * 32 + 16 + hi * 8];
        v16h af = cat16(a0, a1);
#pragma unroll
        for (int nt = 0; nt < 4; nt++) {
            int nl = nt * 16 + lo;
            h8 b0 = *(const h8*)&lB[nl * 32 + hi * 8];
            h8 b1 = *(const h8*)&lB[nl * 32 + 16 + hi * 8];
            v16h bf = cat16(b0, b1);
            acc[nt] = __builtin_amdgcn_wmma_f32_16x16x32_f16(false, af, false, bf,
                                                             (short)0, acc[nt], false, false);
        }
        __syncthreads();   // done reading lds[cur]; next iter may overwrite it
        cur ^= 1;
    }

    const float inv_sqrt_ck = 0.044194173824159216f;  // 1/sqrt(512)
#pragma unroll
    for (int nt = 0; nt < 4; nt++) {
#pragma unroll
        for (int r = 0; r < 8; r++) {
            int m = m0 + w * 16 + r + 8 * hi;
            int n = n0 + nt * 16 + lo;
            if (mode == 0) {
                out[((size_t)b * M + m) * N + n] =
                    (2.0f * acc[nt][r] - rs[b * M + m]) * inv_sqrt_ck;
            } else {
                out[((size_t)b * Ctot + m) * N + n] = acc[nt][r];
            }
        }
    }
}

// ---------------------------------------------------------------------------
// 3x3 SAME conv as 9 shifted GEMM taps over transposed-f16 activations.
// XT: [B, H*W, Cin] f16 (relu/pool pre-fused at staging time)
// WP: [9, Cout, Cin] f16 (tap-major packed OIHW weights)
// One wave = 16(M) x 64(N) tile; 4 WMMA per K-step, all loads 16B.
// ---------------------------------------------------------------------------
__global__ void k_conv_tap(const _Float16* __restrict__ XT, const _Float16* __restrict__ WP,
                           const float* __restrict__ bias, float* __restrict__ out,
                           int Cin, int Cout, int Hc, int Wc, int sig_out) {
    int lane = threadIdx.x, hi = lane >> 4, lo = lane & 15;
    int n0 = blockIdx.x * 64, m0 = blockIdx.y * 16, b = blockIdx.z;
    int HWc = Hc * Wc;
    const _Float16* Xb = XT + (size_t)b * HWc * Cin;
    v8f acc[4] = {{}, {}, {}, {}};
    v16h zf;
#pragma unroll
    for (int e = 0; e < 16; e++) zf[e] = (_Float16)0.0f;

    for (int tap = 0; tap < 9; tap++) {
        int kh = tap / 3, kw = tap - 3 * kh;
        const _Float16* Wrow = WP + ((size_t)tap * Cout + m0 + lo) * Cin;
        const _Float16* rowp[4];
        bool valid[4];
#pragma unroll
        for (int nt = 0; nt < 4; nt++) {
            int n = n0 + nt * 16 + lo;
            int oy = n / Wc, ox = n % Wc;
            int iy = oy + kh - 1, ix = ox + kw - 1;
            valid[nt] = (iy >= 0 && iy < Hc && ix >= 0 && ix < Wc);
            rowp[nt] = Xb + (size_t)(valid[nt] ? (iy * Wc + ix) : 0) * Cin;
        }
        for (int k0 = 0; k0 < Cin; k0 += 32) {
            h8 a0 = *(const h8*)&Wrow[k0 + hi * 8];
            h8 a1 = *(const h8*)&Wrow[k0 + 16 + hi * 8];
            v16h af = cat16(a0, a1);
#pragma unroll
            for (int nt = 0; nt < 4; nt++) {
                h8 b0 = *(const h8*)&rowp[nt][k0 + hi * 8];
                h8 b1 = *(const h8*)&rowp[nt][k0 + 16 + hi * 8];
                v16h bf = cat16(b0, b1);
                // per-lane zero for padded pixels: lane-level cndmask, EXEC untouched
                bf = valid[nt] ? bf : zf;
                acc[nt] = __builtin_amdgcn_wmma_f32_16x16x32_f16(false, af, false, bf,
                                                                 (short)0, acc[nt], false, false);
            }
        }
    }
#pragma unroll
    for (int nt = 0; nt < 4; nt++) {
#pragma unroll
        for (int r = 0; r < 8; r++) {
            int m = m0 + r + 8 * hi;
            int n = n0 + nt * 16 + lo;
            float v = acc[nt][r] + bias[m];
            if (sig_out) v = 1.0f / (1.0f + __expf(-v));
            out[((size_t)b * Cout + m) * HWc + n] = v;
        }
    }
}

// ---------------------------------------------------------------------------
// Staging / pointwise kernels
// ---------------------------------------------------------------------------
// f32 [B,C,NP] -> f16 [B,NP,C] with optional relu and per-(b,c) scalar add.
__global__ void k_stage_xt(const float* __restrict__ in, _Float16* __restrict__ out,
                           int C, int NP, int relu, const float* __restrict__ addchan) {
    size_t total = (size_t)Bn * NP * C;
    size_t i = (size_t)blockIdx.x * 256 + threadIdx.x;
    if (i >= total) return;
    int c = (int)(i % C);
    size_t t = i / C;
    int n = (int)(t % NP);
    int b = (int)(t / NP);
    float v = in[((size_t)b * C + c) * NP + n];
    if (addchan) v += addchan[b * C + c];
    if (relu) v = fmaxf(v, 0.0f);
    out[i] = (_Float16)v;
}

__global__ void k_cast_h(const float* __restrict__ in, _Float16* __restrict__ out, size_t n) {
    size_t i = (size_t)blockIdx.x * 256 + threadIdx.x;
    if (i < n) out[i] = (_Float16)in[i];
}

// OIHW f32 -> [9][Cout][Cin] f16
__global__ void k_pack_w(const float* __restrict__ w, _Float16* __restrict__ wp,
                         int Cout, int Cin) {
    size_t total = (size_t)Cout * Cin * 9;
    size_t i = (size_t)blockIdx.x * 256 + threadIdx.x;
    if (i >= total) return;
    int ci = (int)(i % Cin);
    size_t t = i / Cin;
    int o = (int)(t % Cout);
    int tap = (int)(t / Cout);
    wp[i] = (_Float16)w[((size_t)o * Cin + ci) * 9 + tap];
}

__global__ void k_rowsumsq(const float* __restrict__ mk, float* __restrict__ rs) {
    int idx = blockIdx.x;
    int b = idx / THW, t = idx % THW;
    const float* p = mk + (size_t)b * CK * THW + t;
    float s = 0.0f;
    for (int c = threadIdx.x; c < CK; c += 32) {
        float v = p[(size_t)c * THW];
        s += v * v;
    }
    for (int off = 16; off > 0; off >>= 1) s += __shfl_down(s, off, 32);
    if (threadIdx.x == 0) rs[idx] = s;
}

// Column softmax over t, writing transposed f16 [B,HW,THW] for the readout GEMM.
__global__ void k_softmax_t(const float* __restrict__ aff, _Float16* __restrict__ afft) {
    int col = blockIdx.x;
    int b = col / HW, n = col % HW;
    const float* base = aff + (size_t)b * THW * HW + n;
    _Float16* obase = afft + ((size_t)b * HW + n) * THW;
    __shared__ float red[256];
    int tid = threadIdx.x;
    float mx = -1e30f;
    for (int t = tid; t < THW; t += 256) mx = fmaxf(mx, base[(size_t)t * HW]);
    red[tid] = mx; __syncthreads();
    for (int s = 128; s > 0; s >>= 1) {
        if (tid < s) red[tid] = fmaxf(red[tid], red[tid + s]);
        __syncthreads();
    }
    mx = red[0]; __syncthreads();
    float sm = 0.0f;
    for (int t = tid; t < THW; t += 256) sm += __expf(base[(size_t)t * HW] - mx);
    red[tid] = sm; __syncthreads();
    for (int s = 128; s > 0; s >>= 1) {
        if (tid < s) red[tid] += red[tid + s];
        __syncthreads();
    }
    float inv = 1.0f / red[0];
    for (int t = tid; t < THW; t += 256)
        obase[t] = (_Float16)(__expf(base[(size_t)t * HW] - mx) * inv);
}

__global__ void k_pool(const float* __restrict__ f, float* __restrict__ pool, int HWc) {
    int idx = blockIdx.x;
    const float* p = f + (size_t)idx * HWc;
    float s = 0.0f;
    for (int i = threadIdx.x; i < HWc; i += 32) s += p[i];
    for (int off = 16; off > 0; off >>= 1) s += __shfl_down(s, off, 32);
    if (threadIdx.x == 0) pool[idx] = s / (float)HWc;
}

__global__ void k_decodes(const float* __restrict__ fusion, const float* __restrict__ mbuf,
                          const float* __restrict__ qv, float* __restrict__ dec) {
    size_t i = (size_t)blockIdx.x * 256 + threadIdx.x;
    size_t total = (size_t)Bn * 1024 * HW;
    if (i >= total) return;
    int n = (int)(i % HW);
    size_t bc = i / HW;
    int c = (int)(bc % 1024);
    int b = (int)(bc / 1024);
    float v;
    if (c < 512) {
        float l = fusion[((size_t)b * 1536 + c) * HW + n];
        float g = fusion[((size_t)b * 1536 + 512 + c) * HW + n];
        float a = fusion[((size_t)b * 1536 + 1024 + c) * HW + n];
        float m = mbuf[((size_t)b * 512 + c) * HW + n];
        v = (l + g + a) * (m + 1.0f);
    } else {
        v = qv[((size_t)b * 512 + (c - 512)) * HW + n];
    }
    dec[i] = v;
}

__global__ void k_add(float* __restrict__ y, const float* __restrict__ x, size_t n) {
    size_t i = (size_t)blockIdx.x * 256 + threadIdx.x;
    if (i < n) y[i] += x[i];
}

__global__ void k_up2_add(float* __restrict__ out, const float* __restrict__ in,
                          int C, int h, int w) {
    int H2 = 2 * h, W2 = 2 * w;
    size_t total = (size_t)Bn * C * H2 * W2;
    size_t i = (size_t)blockIdx.x * 256 + threadIdx.x;
    if (i >= total) return;
    int ox = (int)(i % W2);
    size_t t = i / W2;
    int oy = (int)(t % H2);
    size_t bc = t / H2;
    float sy = fmaxf((oy + 0.5f) * 0.5f - 0.5f, 0.0f);
    float sx = fmaxf((ox + 0.5f) * 0.5f - 0.5f, 0.0f);
    int y0 = (int)sy, x0 = (int)sx;
    int y1 = min(y0 + 1, h - 1), x1 = min(x0 + 1, w - 1);
    float wy = sy - y0, wx = sx - x0;
    const float* p = in + bc * (size_t)h * w;
    float v = (1.0f - wy) * ((1.0f - wx) * p[y0 * w + x0] + wx * p[y0 * w + x1]) +
              wy * ((1.0f - wx) * p[y1 * w + x0] + wx * p[y1 * w + x1]);
    out[i] += v;
}

__global__ void k_pred(const float* __restrict__ in, const float* __restrict__ w,
                       const float* __restrict__ bias, float* __restrict__ out,
                       int Cin, int Hc, int Wc) {
    size_t total = (size_t)Bn * Hc * Wc;
    size_t i = (size_t)blockIdx.x * 256 + threadIdx.x;
    if (i >= total) return;
    int x = (int)(i % Wc);
    size_t t = i / Wc;
    int y = (int)(t % Hc);
    int b = (int)(t / Hc);
    float s = bias[0];
    const float* inb = in + (size_t)b * Cin * Hc * Wc;
    for (int c = 0; c < Cin; c++) {
        for (int kh = 0; kh < 3; kh++) {
            int iy = y + kh - 1;
            if (iy < 0 || iy >= Hc) continue;
            for (int kw = 0; kw < 3; kw++) {
                int ix = x + kw - 1;
                if (ix < 0 || ix >= Wc) continue;
                s += fmaxf(inb[(size_t)c * Hc * Wc + iy * Wc + ix], 0.0f) *
                     w[(c * 3 + kh) * 3 + kw];
            }
        }
    }
    out[i] = s;
}

__global__ void k_final(const float* __restrict__ pred96, float* __restrict__ logits,
                        float* __restrict__ prob) {
    const int HO = 384, WO = 384, hi2 = 96, wi2 = 96;
    size_t total = (size_t)Bn * HO * WO;
    size_t i = (size_t)blockIdx.x * 256 + threadIdx.x;
    if (i >= total) return;
    int ox = (int)(i % WO);
    size_t t = i / WO;
    int oy = (int)(t % HO);
    int b = (int)(t / HO);
    float sy = fmaxf((oy + 0.5f) * 0.25f - 0.5f, 0.0f);
    float sx = fmaxf((ox + 0.5f) * 0.25f - 0.5f, 0.0f);
    int y0 = (int)sy, x0 = (int)sx;
    int y1 = min(y0 + 1, hi2 - 1), x1 = min(x0 + 1, wi2 - 1);
    float wy = sy - y0, wx = sx - x0;
    const float* p = pred96 + (size_t)b * hi2 * wi2;
    float v = (1.0f - wy) * ((1.0f - wx) * p[y0 * wi2 + x0] + wx * p[y0 * wi2 + x1]) +
              wy * ((1.0f - wx) * p[y1 * wi2 + x0] + wx * p[y1 * wi2 + x1]);
    float pr = 1.0f / (1.0f + __expf(-v));
    float bg = 1.0f - pr;
    float p0 = fminf(fmaxf(bg, 1e-7f), 1.0f - 1e-7f);
    float p1 = fminf(fmaxf(pr, 1e-7f), 1.0f - 1e-7f);
    float l0 = __logf(p0 / (1.0f - p0));
    float l1 = __logf(p1 / (1.0f - p1));
    size_t base = ((size_t)b * 2) * HO * WO + (size_t)oy * WO + ox;
    logits[base] = l0;
    logits[base + (size_t)HO * WO] = l1;
    float mx = fmaxf(l0, l1);
    float e0 = __expf(l0 - mx), e1 = __expf(l1 - mx);
    prob[(size_t)b * HO * WO + (size_t)oy * WO + ox] = e1 / (e0 + e1);
}

// ---------------------------------------------------------------------------
// Launcher
// ---------------------------------------------------------------------------
static inline unsigned cdiv(size_t a, unsigned b) { return (unsigned)((a + b - 1) / b); }

extern "C" void kernel_launch(void* const* d_in, const int* in_sizes, int n_in,
                              void* d_out, int out_size, void* d_ws, size_t ws_size,
                              hipStream_t stream) {
    (void)in_sizes; (void)n_in; (void)out_size; (void)ws_size;

    const float* qk16 = (const float*)d_in[0];
    const float* qv16 = (const float*)d_in[1];
    const float* qf8  = (const float*)d_in[2];
    const float* qf4  = (const float*)d_in[3];
    const float* keys[3] = { (const float*)d_in[4], (const float*)d_in[6], (const float*)d_in[8] };
    const float* vals[3] = { (const float*)d_in[5], (const float*)d_in[7], (const float*)d_in[9] };
    const float* mask_w    = (const float*)d_in[10];
    const float* mask_b    = (const float*)d_in[11];
    const float* cmp_ds_w  = (const float*)d_in[12];
    const float* cmp_ds_b  = (const float*)d_in[13];
    const float* cmp_c1_w  = (const float*)d_in[14];
    const float* cmp_c1_b  = (const float*)d_in[15];
    const float* cmp_c2_w  = (const float*)d_in[16];
    const float* cmp_c2_b  = (const float*)d_in[17];
    const float* u1_skip_w = (const float*)d_in[18];
    const float* u1_skip_b = (const float*)d_in[19];
    const float* u1_ds_w   = (const float*)d_in[20];
    const float* u1_ds_b   = (const float*)d_in[21];
    const float* u1_c1_w   = (const float*)d_in[22];
    const float* u1_c1_b   = (const float*)d_in[23];
    const float* u1_c2_w   = (const float*)d_in[24];
    const float* u1_c2_b   = (const float*)d_in[25];
    const float* u2_skip_w = (const float*)d_in[26];
    const float* u2_skip_b = (const float*)d_in[27];
    const float* u2_c1_w   = (const float*)d_in[28];
    const float* u2_c1_b   = (const float*)d_in[29];
    const float* u2_c2_w   = (const float*)d_in[30];
    const float* u2_c2_b   = (const float*)d_in[31];
    const float* pred_w    = (const float*)d_in[32];
    const float* pred_b    = (const float*)d_in[33];

    float* ws = (float*)d_ws;
    float* AFF    = ws + OFF_AFF;
    float* ROWSQ  = ws + OFF_ROWSQ;
    float* FUSION = ws + OFF_FUSION;
    float* POOL   = ws + OFF_POOL;
    float* MBUF   = ws + OFF_MBUF;
    float* DEC    = ws + OFF_DEC;
    float* T1     = ws + OFF_T1;
    float* T2     = ws + OFF_T2;
    float* X16    = ws + OFF_X16;
    float* S8     = ws + OFF_S8;
    float* X8     = ws + OFF_X8;
    float* S4     = ws + OFF_S4;
    float* PRED   = ws + OFF_PRED;

    _Float16* hb   = (_Float16*)(ws + F32_TOTAL);
    _Float16* KT   = hb + HOFF_KT;
    _Float16* QT   = hb + HOFF_QT;
    _Float16* MVH  = hb + HOFF_MVH;
    _Float16* AFFT = hb + HOFF_AFFT;
    _Float16* WP   = hb + HOFF_WP;
    _Float16* XT1  = hb + HOFF_XT1;
    _Float16* XT2  = hb + HOFF_XT2;

    // ---- Stage 0: stage qk^T once ----
    k_stage_xt<<<cdiv((size_t)Bn * HW * CK, 256), 256, 0, stream>>>(qk16, QT, CK, HW, 0, nullptr);

    // ---- Stage 1: attention per bank; readout lands in FUSION slices ----
    for (int j = 0; j < 3; j++) {
        k_stage_xt<<<cdiv(HZ_KT, 256), 256, 0, stream>>>(keys[j], KT, CK, THW, 0, nullptr);
        k_cast_h<<<cdiv(HZ_MVH, 256), 256, 0, stream>>>(vals[j], MVH, HZ_MVH);
        k_rowsumsq<<<dim3(Bn * THW), dim3(32), 0, stream>>>(keys[j], ROWSQ);
        k_gemm64<<<dim3(HW / 64, THW / 64, Bn), dim3(128), 0, stream>>>(
            KT, QT, ROWSQ, AFF, THW, HW, CK, 0, 0);
        k_softmax_t<<<dim3(Bn * HW), dim3(256), 0, stream>>>(AFF, AFFT);
        k_gemm64<<<dim3(HW / 64, CV / 64, Bn), dim3(128), 0, stream>>>(
            MVH, AFFT, ROWSQ, FUSION + (size_t)j * 512 * HW, CV, HW, THW, 1, 1536);
    }

    // ---- Stage 2: pool + mask conv + decodes ----
    k_pool<<<dim3(Bn * 1536), dim3(32), 0, stream>>>(FUSION, POOL, HW);
    k_stage_xt<<<cdiv((size_t)Bn * HW * 1536, 256), 256, 0, stream>>>(FUSION, XT1, 1536, HW, 0, POOL);
    k_pack_w<<<cdiv((size_t)512 * 1536 * 9, 256), 256, 0, stream>>>(mask_w, WP, 512, 1536);
    k_conv_tap<<<dim3(HW / 64, 512 / 16, Bn), dim3(32), 0, stream>>>(
        XT1, WP, mask_b, MBUF, 1536, 512, Hh, Ww, 1);
    k_decodes<<<cdiv((size_t)Bn * 1024 * HW, 256), 256, 0, stream>>>(FUSION, MBUF, qv16, DEC);

    // ---- Stage 3: cmp resblock @24x24 ----
    k_stage_xt<<<cdiv((size_t)Bn * HW * 1024, 256), 256, 0, stream>>>(DEC, XT1, 1024, HW, 1, nullptr);
    k_stage_xt<<<cdiv((size_t)Bn * HW * 1024, 256), 256, 0, stream>>>(DEC, XT2, 1024, HW, 0, nullptr);
    k_pack_w<<<cdiv((size_t)512 * 1024 * 9, 256), 256, 0, stream>>>(cmp_c1_w, WP, 512, 1024);
    k_conv_tap<<<dim3(HW / 64, 512 / 16, Bn), dim3(32), 0, stream>>>(
        XT1, WP, cmp_c1_b, T1, 1024, 512, Hh, Ww, 0);
    k_pack_w<<<cdiv((size_t)512 * 1024 * 9, 256), 256, 0, stream>>>(cmp_ds_w, WP, 512, 1024);
    k_conv_tap<<<dim3(HW / 64, 512 / 16, Bn), dim3(32), 0, stream>>>(
        XT2, WP, cmp_ds_b, X16, 1024, 512, Hh, Ww, 0);
    k_stage_xt<<<cdiv((size_t)Bn * HW * 512, 256), 256, 0, stream>>>(T1, XT1, 512, HW, 1, nullptr);
    k_pack_w<<<cdiv((size_t)512 * 512 * 9, 256), 256, 0, stream>>>(cmp_c2_w, WP, 512, 512);
    k_conv_tap<<<dim3(HW / 64, 512 / 16, Bn), dim3(32), 0, stream>>>(
        XT1, WP, cmp_c2_b, T2, 512, 512, Hh, Ww, 0);
    k_add<<<cdiv((size_t)Bn * 512 * HW, 256), 256, 0, stream>>>(X16, T2, (size_t)Bn * 512 * HW);

    // ---- Stage 4: u1 @48x48 ----
    k_stage_xt<<<cdiv((size_t)Bn * 2304 * 512, 256), 256, 0, stream>>>(qf8, XT1, 512, 2304, 0, nullptr);
    k_pack_w<<<cdiv((size_t)512 * 512 * 9, 256), 256, 0, stream>>>(u1_skip_w, WP, 512, 512);
    k_conv_tap<<<dim3(2304 / 64, 512 / 16, Bn), dim3(32), 0, stream>>>(
        XT1, WP, u1_skip_b, S8, 512, 512, 48, 48, 0);
    k_up2_add<<<cdiv((size_t)Bn * 512 * 2304, 256), 256, 0, stream>>>(S8, X16, 512, 24, 24);
    k_stage_xt<<<cdiv((size_t)Bn * 2304 * 512, 256), 256, 0, stream>>>(S8, XT1, 512, 2304, 1, nullptr);
    k_stage_xt<<<cdiv((size_t)Bn * 2304 * 512, 256), 256, 0, stream>>>(S8, XT2, 512, 2304, 0, nullptr);
    k_pack_w<<<cdiv((size_t)256 * 512 * 9, 256), 256, 0, stream>>>(u1_c1_w, WP, 256, 512);
    k_conv_tap<<<dim3(2304 / 64, 256 / 16, Bn), dim3(32), 0, stream>>>(
        XT1, WP, u1_c1_b, T1, 512, 256, 48, 48, 0);
    k_pack_w<<<cdiv((size_t)256 * 512 * 9, 256), 256, 0, stream>>>(u1_ds_w, WP, 256, 512);
    k_conv_tap<<<dim3(2304 / 64, 256 / 16, Bn), dim3(32), 0, stream>>>(
        XT2, WP, u1_ds_b, X8, 512, 256, 48, 48, 0);
    k_stage_xt<<<cdiv((size_t)Bn * 2304 * 256, 256), 256, 0, stream>>>(T1, XT1, 256, 2304, 1, nullptr);
    k_pack_w<<<cdiv((size_t)256 * 256 * 9, 256), 256, 0, stream>>>(u1_c2_w, WP, 256, 256);
    k_conv_tap<<<dim3(2304 / 64, 256 / 16, Bn), dim3(32), 0, stream>>>(
        XT1, WP, u1_c2_b, T2, 256, 256, 48, 48, 0);
    k_add<<<cdiv((size_t)Bn * 256 * 2304, 256), 256, 0, stream>>>(X8, T2, (size_t)Bn * 256 * 2304);

    // ---- Stage 5: u2 @96x96 + pred ----
    k_stage_xt<<<cdiv((size_t)Bn * 9216 * 256, 256), 256, 0, stream>>>(qf4, XT1, 256, 9216, 0, nullptr);
    k_pack_w<<<cdiv((size_t)256 * 256 * 9, 256), 256, 0, stream>>>(u2_skip_w, WP, 256, 256);
    k_conv_tap<<<dim3(9216 / 64, 256 / 16, Bn), dim3(32), 0, stream>>>(
        XT1, WP, u2_skip_b, S4, 256, 256, 96, 96, 0);
    k_up2_add<<<cdiv((size_t)Bn * 256 * 9216, 256), 256, 0, stream>>>(S4, X8, 256, 48, 48);
    k_stage_xt<<<cdiv((size_t)Bn * 9216 * 256, 256), 256, 0, stream>>>(S4, XT1, 256, 9216, 1, nullptr);
    k_pack_w<<<cdiv((size_t)256 * 256 * 9, 256), 256, 0, stream>>>(u2_c1_w, WP, 256, 256);
    k_conv_tap<<<dim3(9216 / 64, 256 / 16, Bn), dim3(32), 0, stream>>>(
        XT1, WP, u2_c1_b, T1, 256, 256, 96, 96, 0);
    k_stage_xt<<<cdiv((size_t)Bn * 9216 * 256, 256), 256, 0, stream>>>(T1, XT2, 256, 9216, 1, nullptr);
    k_pack_w<<<cdiv((size_t)256 * 256 * 9, 256), 256, 0, stream>>>(u2_c2_w, WP, 256, 256);
    k_conv_tap<<<dim3(9216 / 64, 256 / 16, Bn), dim3(32), 0, stream>>>(
        XT2, WP, u2_c2_b, T2, 256, 256, 96, 96, 0);
    k_add<<<cdiv((size_t)Bn * 256 * 9216, 256), 256, 0, stream>>>(S4, T2, (size_t)Bn * 256 * 9216);
    k_pred<<<cdiv((size_t)Bn * 9216, 256), 256, 0, stream>>>(S4, pred_w, pred_b, PRED, 256, 96, 96);

    // ---- Stage 6: resize x4 + sigmoid + aggregate ----
    float* out_logits = (float*)d_out;
    float* out_prob   = out_logits + (size_t)Bn * 2 * 384 * 384;
    k_final<<<cdiv((size_t)Bn * 384 * 384, 256), 256, 0, stream>>>(PRED, out_logits, out_prob);
}
